// PSAttention_6476810683003
// MI455X (gfx1250) — matrix operand order repaired and checked by reading, not compiled, not run on hardware
//
#include <hip/hip_runtime.h>

// ---------------------------------------------------------------------------
// Problem constants (reference: B=32, CIN=2048, COUT=1024, H=W=32, HW=1024)
// ---------------------------------------------------------------------------
#define BATCH 32
#define CIN   2048
#define CDIM  1024   // COUT == H*W == 1024

typedef __attribute__((ext_vector_type(16))) __bf16 v16bf;
typedef __attribute__((ext_vector_type(8)))  float  v8f;
// payload type expected by the async-LDS builtins (clang vector, not HIP int4)
typedef __attribute__((__vector_size__(4 * sizeof(int)))) int v4i;

union FragAB { uint4 u4[2]; v16bf v; };

// CDNA5 async global->LDS path (guarded: falls back to register staging)
#if defined(__gfx1250__) && \
    __has_builtin(__builtin_amdgcn_global_load_async_to_lds_b128) && \
    __has_builtin(__builtin_amdgcn_s_wait_asynccnt)
#define USE_ASYNC_LDS 1
#define AS_GLOBAL __attribute__((address_space(1)))
#define AS_LOCAL  __attribute__((address_space(3)))
#else
#define USE_ASYNC_LDS 0
#endif

// round-to-nearest-even f32 -> bf16 (raw bits)
static __device__ __forceinline__ unsigned short f2bf(float f) {
    unsigned u = __float_as_uint(f);
    u += 0x7fffu + ((u >> 16) & 1u);
    return (unsigned short)(u >> 16);
}

// ---------------------------------------------------------------------------
// f32 -> bf16 bulk conversion (vectorized: float4 in, 4x bf16 packed u64 out)
// ---------------------------------------------------------------------------
__global__ __launch_bounds__(256)
void convert_f32_to_bf16(const float* __restrict__ in,
                         unsigned short* __restrict__ out, size_t n4) {
    size_t i = (size_t)blockIdx.x * blockDim.x + threadIdx.x;
    if (i >= n4) return;
    float4 f = ((const float4*)in)[i];
    unsigned long long pk =
        (unsigned long long)f2bf(f.x)
      | ((unsigned long long)f2bf(f.y) << 16)
      | ((unsigned long long)f2bf(f.z) << 32)
      | ((unsigned long long)f2bf(f.w) << 48);
    ((unsigned long long*)out)[i] = pk;
}

// ---------------------------------------------------------------------------
// Tiled bf16 WMMA GEMM:  C[b] = A[b](MxK) * B[b](KxN) (+bias[m]) (+resid)
//   - 256 threads = 8 waves (wave32), 128x128 block tile, K-step 32
//   - each wave: 64x32 sub-tile = 4x2 v_wmma_f32_16x16x32_bf16 per K-step
//   - double-buffered LDS; async global->LDS (ASYNCcnt) when available
//   - optional transposed f32 store (for the column-softmax branch)
//   - optional fused bf16 secondary output (for Y reuse as later B operand)
// A/B are raw-bf16 (unsigned short) buffers; strides are in elements.
// ---------------------------------------------------------------------------
__global__ __launch_bounds__(256)
void gemm_bf16_wmma(const unsigned short* __restrict__ A, long long strideA,
                    const unsigned short* __restrict__ Bm, long long strideB,
                    const float* __restrict__ bias,
                    const float* __restrict__ resid, long long strideR,
                    float* __restrict__ Cf, long long strideC,
                    unsigned short* __restrict__ Cb, long long strideCb,
                    int M, int N, int K, int transposeOut)
{
    // Ping-pong LDS tiles, padded to dodge bank conflicts (rows 16B-aligned)
    __shared__ __align__(16) unsigned short sA[2][128 * 40];  // 128x32 (+8 pad)
    __shared__ __align__(16) unsigned short sB[2][32 * 136];  // 32x128 (+8 pad)

    const int tid   = threadIdx.x;
    const int lane  = tid & 31;
    const int w     = tid >> 5;
    const int waveR = w >> 2;           // 0..1  (M direction, 64 rows each)
    const int waveC = w & 3;            // 0..3  (N direction, 32 cols each)
    const int batch = blockIdx.z;
    const int mBlock = blockIdx.y * 128;
    const int nBlock = blockIdx.x * 128;

    A  += (size_t)batch * strideA;
    Bm += (size_t)batch * strideB;

    v8f acc[4][2];
    {
        v8f zero = {0.f, 0.f, 0.f, 0.f, 0.f, 0.f, 0.f, 0.f};
        #pragma unroll
        for (int i = 0; i < 4; ++i)
            #pragma unroll
            for (int j = 0; j < 2; ++j) acc[i][j] = zero;
    }

    // staging assignments: each thread moves 32B of A and 32B of B per K-step
    const int aRow = tid >> 1, aSeg = (tid & 1) << 4;   // 128 rows x 32 halfs
    const int bRow = tid >> 3, bSeg = (tid & 7) << 4;   // 32 rows x 128 halfs
    const int khalf = lane >> 4;                        // A-frag K-half select

    // --- compute from one LDS buffer: 8 WMMAs -------------------------------
    auto compute = [&](int buf) {
        // A fragments: ISA 16-bit A 16x32 layout — lanes 0-15 hold K0-7 &
        // K16-23, lanes 16-31 hold K8-15 & K24-31 (two 16B chunks per lane)
        FragAB fa[4], fb[2];
        #pragma unroll
        for (int i = 0; i < 4; ++i) {
            const uint4* rp =
                (const uint4*)&sA[buf][(waveR * 64 + i * 16 + (lane & 15)) * 40];
            fa[i].u4[0] = rp[khalf];
            fa[i].u4[1] = rp[khalf + 2];
        }
        // B fragments: 32x16 — lane holds column n = lane&15, K in same order as A
        #pragma unroll
        for (int j = 0; j < 2; ++j) {
            const unsigned short* cp =
                &sB[buf][waveC * 32 + j * 16 + (lane & 15)];
            unsigned short* d = (unsigned short*)&fb[j];
            #pragma unroll
            for (int s = 0; s < 16; ++s) {
                const int k = (s < 8) ? (khalf * 8 + s) : (16 + khalf * 8 + (s - 8));
                d[s] = cp[k * 136];
            }
        }
        #pragma unroll
        for (int i = 0; i < 4; ++i)
            #pragma unroll
            for (int j = 0; j < 2; ++j)
                acc[i][j] = __builtin_amdgcn_wmma_f32_16x16x32_bf16(
                    false, fa[i].v, false, fb[j].v,
                    (short)0, acc[i][j], false, false);
    };

    const int nIt = K >> 5;

#if USE_ASYNC_LDS
    // --- async global->LDS staging: 4x B128, tracked by ASYNCcnt -----------
    auto stage_async = [&](int buf, int k0) {
        const unsigned short* ga = A + (size_t)(mBlock + aRow) * K + k0 + aSeg;
        unsigned short*       la = &sA[buf][aRow * 40 + aSeg];
        __builtin_amdgcn_global_load_async_to_lds_b128(
            (AS_GLOBAL v4i*)ga, (AS_LOCAL v4i*)la, 0, 0);
        __builtin_amdgcn_global_load_async_to_lds_b128(
            (AS_GLOBAL v4i*)(ga + 8), (AS_LOCAL v4i*)(la + 8), 0, 0);
        const unsigned short* gb = Bm + (size_t)(k0 + bRow) * N + nBlock + bSeg;
        unsigned short*       lb = &sB[buf][bRow * 136 + bSeg];
        __builtin_amdgcn_global_load_async_to_lds_b128(
            (AS_GLOBAL v4i*)gb, (AS_LOCAL v4i*)lb, 0, 0);
        __builtin_amdgcn_global_load_async_to_lds_b128(
            (AS_GLOBAL v4i*)(gb + 8), (AS_LOCAL v4i*)(lb + 8), 0, 0);
    };

    stage_async(0, 0);
    for (int it = 0; it < nIt; ++it) {
        const int cur = it & 1;
        const bool have = (it + 1 < nIt);
        if (have) stage_async(cur ^ 1, (it + 1) << 5);
        // async loads complete in order: <=4 pending means current tile landed
        if (have) __builtin_amdgcn_s_wait_asynccnt(4);
        else      __builtin_amdgcn_s_wait_asynccnt(0);
        __syncthreads();            // current buffer visible to all waves
        compute(cur);
        __syncthreads();            // done reading before next overwrite
    }
#else
    // --- register-staged double buffer (one barrier per K-step) ------------
    auto gload = [&](int k0, uint4& ra0, uint4& ra1, uint4& rb0, uint4& rb1) {
        const uint4* gpA = (const uint4*)(A + (size_t)(mBlock + aRow) * K + k0 + aSeg);
        ra0 = gpA[0]; ra1 = gpA[1];
        const uint4* gpB = (const uint4*)(Bm + (size_t)(k0 + bRow) * N + nBlock + bSeg);
        rb0 = gpB[0]; rb1 = gpB[1];
    };
    auto sstore = [&](int buf, uint4 ra0, uint4 ra1, uint4 rb0, uint4 rb1) {
        uint4* spA = (uint4*)&sA[buf][aRow * 40 + aSeg];
        spA[0] = ra0; spA[1] = ra1;
        uint4* spB = (uint4*)&sB[buf][bRow * 136 + bSeg];
        spB[0] = rb0; spB[1] = rb1;
    };

    {
        uint4 ra0, ra1, rb0, rb1;
        gload(0, ra0, ra1, rb0, rb1);
        sstore(0, ra0, ra1, rb0, rb1);
    }
    for (int it = 0; it < nIt; ++it) {
        const int cur = it & 1;
        const bool have = (it + 1 < nIt);
        uint4 ra0, ra1, rb0, rb1;
        if (have) gload((it + 1) << 5, ra0, ra1, rb0, rb1);
        __syncthreads();            // cur buffer's stores visible to all waves
        compute(cur);
        if (have) sstore(cur ^ 1, ra0, ra1, rb0, rb1);  // fill idle buffer
    }
#endif

    // epilogue: C/D layout -> m = r + 8*(lane>=16), n = lane&15 per 16x16 tile
    Cf += (size_t)batch * strideC;
    if (Cb)    Cb    += (size_t)batch * strideCb;
    if (resid) resid += (size_t)batch * strideR;

    const int mW = mBlock + waveR * 64;
    const int nW = nBlock + waveC * 32;
    #pragma unroll
    for (int i = 0; i < 4; ++i) {
        #pragma unroll
        for (int j = 0; j < 2; ++j) {
            #pragma unroll
            for (int r = 0; r < 8; ++r) {
                const int m = mW + i * 16 + (lane >> 4) * 8 + r;
                const int n = nW + j * 16 + (lane & 15);
                float v = acc[i][j][r];
                if (bias)  v += bias[m];
                if (resid) v += resid[(size_t)m * N + n];
                if (transposeOut) {
                    Cf[(size_t)n * M + m] = v;
                } else {
                    Cf[(size_t)m * N + n] = v;
                    if (Cb) Cb[(size_t)m * N + n] = f2bf(v);
                }
            }
        }
    }
}

// ---------------------------------------------------------------------------
// Row softmax (f32 in -> bf16 out), one 256-thread block per 1024-elem row.
// ---------------------------------------------------------------------------
__global__ __launch_bounds__(256)
void row_softmax_bf16(const float* __restrict__ in,
                      unsigned short* __restrict__ out, int cols)
{
    const size_t row = blockIdx.x;
    const float* p = in + row * (size_t)cols;
    const int tid = threadIdx.x, lane = tid & 31, wid = tid >> 5;
    __shared__ float sred[8];

    float v[4];
    #pragma unroll
    for (int i = 0; i < 4; ++i) v[i] = p[tid + (i << 8)];

    float mx = fmaxf(fmaxf(v[0], v[1]), fmaxf(v[2], v[3]));
    #pragma unroll
    for (int off = 16; off > 0; off >>= 1) mx = fmaxf(mx, __shfl_xor(mx, off, 32));
    if (lane == 0) sred[wid] = mx;
    __syncthreads();
    if (tid == 0) {
        float m = sred[0];
        for (int i = 1; i < 8; ++i) m = fmaxf(m, sred[i]);
        sred[0] = m;
    }
    __syncthreads();
    mx = sred[0];
    __syncthreads();

    float e[4], s = 0.f;
    #pragma unroll
    for (int i = 0; i < 4; ++i) { e[i] = __expf(v[i] - mx); s += e[i]; }
    #pragma unroll
    for (int off = 16; off > 0; off >>= 1) s += __shfl_xor(s, off, 32);
    if (lane == 0) sred[wid] = s;
    __syncthreads();
    if (tid == 0) {
        float t = 0.f;
        for (int i = 0; i < 8; ++i) t += sred[i];
        sred[0] = t;
    }
    __syncthreads();
    const float inv = 1.0f / sred[0];

    #pragma unroll
    for (int i = 0; i < 4; ++i)
        out[row * (size_t)cols + tid + (i << 8)] = f2bf(e[i] * inv);
}

// ---------------------------------------------------------------------------
// Host-side orchestration
// ---------------------------------------------------------------------------
extern "C" void kernel_launch(void* const* d_in, const int* in_sizes, int n_in,
                              void* d_out, int out_size, void* d_ws, size_t ws_size,
                              hipStream_t stream) {
    (void)in_sizes; (void)n_in; (void)out_size; (void)ws_size;

    const float* x            = (const float*)d_in[0];  // [B, CIN, H, W]
    const float* w_reduce     = (const float*)d_in[1];  // [CDIM, CIN]
    const float* b_reduce     = (const float*)d_in[2];  // [CDIM]
    const float* w_collect    = (const float*)d_in[3];  // [CDIM, CDIM]
    const float* b_collect    = (const float*)d_in[4];
    const float* w_distribute = (const float*)d_in[5];
    const float* b_distribute = (const float*)d_in[6];
    float* out = (float*)d_out;                          // [B, CDIM, H, W] f32

    // workspace layout (256B aligned chunks)
    char* ws = (char*)d_ws;
    size_t off = 0;
    auto take = [&](size_t bytes) { void* p = ws + off; off += (bytes + 255) & ~(size_t)255; return p; };

    const size_t nX  = (size_t)BATCH * CIN * CDIM;      // 67,108,864
    const size_t nYZ = (size_t)BATCH * CDIM * CDIM;     // 33,554,432

    unsigned short* Xbf  = (unsigned short*)take(nX * 2);          // bf16 x
    unsigned short* Wr   = (unsigned short*)take((size_t)CDIM * CIN * 2);
    unsigned short* Wc   = (unsigned short*)take((size_t)CDIM * CDIM * 2);
    unsigned short* Wd   = (unsigned short*)take((size_t)CDIM * CDIM * 2);
    float*          Yf   = (float*)take(nYZ * 4);                  // reduce out f32
    unsigned short* Ybf  = (unsigned short*)take(nYZ * 2);         // reduce out bf16
    float*          Z    = (float*)take(nYZ * 4);                  // pre-softmax scratch (reused)
    unsigned short* Cbf  = (unsigned short*)take(nYZ * 2);         // softmax(collect)
    unsigned short* Dtbf = (unsigned short*)take(nYZ * 2);         // softmax(distribute)^T

    // 1) f32 -> bf16 conversions
    convert_f32_to_bf16<<<(unsigned)(nX / 4 / 256), 256, 0, stream>>>(x, Xbf, nX / 4);
    convert_f32_to_bf16<<<(unsigned)((size_t)CDIM * CIN / 4 / 256), 256, 0, stream>>>(w_reduce, Wr, (size_t)CDIM * CIN / 4);
    convert_f32_to_bf16<<<(unsigned)((size_t)CDIM * CDIM / 4 / 256), 256, 0, stream>>>(w_collect, Wc, (size_t)CDIM * CDIM / 4);
    convert_f32_to_bf16<<<(unsigned)((size_t)CDIM * CDIM / 4 / 256), 256, 0, stream>>>(w_distribute, Wd, (size_t)CDIM * CDIM / 4);

    const dim3 blk(256);
    const dim3 grd(CDIM / 128, CDIM / 128, BATCH);
    const long long sX = (long long)CIN * CDIM;
    const long long sY = (long long)CDIM * CDIM;

    // 2) Y = W_reduce * X + b_reduce  (f32 + fused bf16 copy)
    gemm_bf16_wmma<<<grd, blk, 0, stream>>>(Wr, 0, Xbf, sX, b_reduce,
                                            nullptr, 0, Yf, sY, Ybf, sY,
                                            CDIM, CDIM, CIN, 0);

    // 3) Z2 = W_collect * Y + b_collect ; row softmax over hw -> Cbf
    gemm_bf16_wmma<<<grd, blk, 0, stream>>>(Wc, 0, Ybf, sY, b_collect,
                                            nullptr, 0, Z, sY, nullptr, 0,
                                            CDIM, CDIM, CDIM, 0);
    row_softmax_bf16<<<BATCH * CDIM, 256, 0, stream>>>(Z, Cbf, CDIM);

    // 4) Z3t = (W_distribute * Y + b_distribute)^T  (transposed store so the
    //    channel-axis softmax becomes a row softmax) -> Dtbf
    gemm_bf16_wmma<<<grd, blk, 0, stream>>>(Wd, 0, Ybf, sY, b_distribute,
                                            nullptr, 0, Z, sY, nullptr, 0,
                                            CDIM, CDIM, CDIM, 1);
    row_softmax_bf16<<<BATCH * CDIM, 256, 0, stream>>>(Z, Dtbf, CDIM);

    // 5) out = Y + Cbf(c,n) * Dtbf(n,d)   (fused residual)
    gemm_bf16_wmma<<<grd, blk, 0, stream>>>(Cbf, sY, Dtbf, sY, nullptr,
                                            Yf, sY, out, sY, nullptr, 0,
                                            CDIM, CDIM, CDIM, 0);
}
